// CAWN_25640954757945
// MI455X (gfx1250) — matrix operand order repaired and checked by reading, not compiled, not run on hardware
//
#include <hip/hip_runtime.h>
#include <hip/hip_bf16.h>
#include <math.h>

// ---------------------------------------------------------------------------
// Shapes (compile-time constants from the reference)
// ---------------------------------------------------------------------------
#define B_TOT 2048
#define NN    64      // neighbors
#define DM    512     // d_model
#define NH    4       // heads
#define DK    128     // d_k
#define FEATD 128     // feat dim
#define XDIM  (DM + FEATD)  // 640, merge-layer input

typedef __attribute__((ext_vector_type(16))) __bf16          v16bf;
typedef __attribute__((ext_vector_type(8)))  float           v8f;
typedef __attribute__((ext_vector_type(8)))  unsigned short  us8;

union FragAB { v16bf v; unsigned short u[16]; us8 h[2]; };

// f32 -> bf16 round-to-nearest-even
__device__ __forceinline__ unsigned short f2bf(float f) {
  unsigned u = __float_as_uint(f);
  unsigned r = u + 0x7FFFu + ((u >> 16) & 1u);
  return (unsigned short)(r >> 16);
}
__device__ __forceinline__ float bf2f(unsigned short h) {
  return __uint_as_float(((unsigned)h) << 16);
}

__device__ __forceinline__ float wred(float v) {
  #pragma unroll
  for (int o = 16; o > 0; o >>= 1) v += __shfl_xor(v, o, 32);
  return v;
}
__device__ __forceinline__ float wmaxr(float v) {
  #pragma unroll
  for (int o = 16; o > 0; o >>= 1) v = fmaxf(v, __shfl_xor(v, o, 32));
  return v;
}

// ---------------------------------------------------------------------------
// WMMA fragment loaders (CDNA5 16x16x32 bf16, ISA 7.12.2 layouts)
//   A (16xK tile 32): lane L -> m = L&15, k = 8*(L>>4) + (j<8 ? j : j+8)
//   B (Kx16 tile 32): lane L -> n = L&15, k = 8*(L>>4) + (j<8 ? j : j+8)
// All operands are bf16 in [row][K] layout so frags are contiguous b128 loads.
// ---------------------------------------------------------------------------
__device__ __forceinline__ v16bf load_a_bf16(const unsigned short* base, int ldk, int kt) {
  const int L  = threadIdx.x & 31;
  const int m  = L & 15;
  const int kb = kt * 32 + ((L >> 4) * 8);
  FragAB f;
  f.h[0] = *((const us8*)(base + (long)m * ldk + kb));        // k = kb..kb+7
  f.h[1] = *((const us8*)(base + (long)m * ldk + kb + 16));   // k = kb+16..kb+23
  return f.v;
}

__device__ __forceinline__ v16bf load_b_wt(const unsigned short* WT, int ldk, int ntile, int kt) {
  const int L  = threadIdx.x & 31;
  const int n  = ntile * 16 + (L & 15);
  const int kb = kt * 32 + ((L >> 4) * 8);
  FragAB f;
  f.h[0] = *((const us8*)(WT + (long)n * ldk + kb));
  f.h[1] = *((const us8*)(WT + (long)n * ldk + kb + 16));
  return f.v;
}

// ---------------------------------------------------------------------------
// Kernel 0: fold w_map into Wq/Wk; transpose+bf16-convert Wv/fc_w/fc1_w/fc2_w
// ---------------------------------------------------------------------------
__global__ void prep_kernel(const float* __restrict__ Wq, const float* __restrict__ Wk,
                            const float* __restrict__ Wv, const float* __restrict__ w_map,
                            const float* __restrict__ fc_w, const float* __restrict__ fc1_w,
                            const float* __restrict__ fc2_w,
                            float* __restrict__ wq_eff, float* __restrict__ wk_eff,
                            unsigned short* __restrict__ WvT, unsigned short* __restrict__ fcT,
                            unsigned short* __restrict__ fc1T, unsigned short* __restrict__ fc2T) {
  const long NEFF = 4096;
  const long NWV  = 512 * 512;
  const long NFC  = 512 * 512;
  const long NF1  = 640 * 128;
  const long NF2  = 128 * 128;
  const long TOT  = NEFF + NWV + NFC + NF1 + NF2;
  const long stride = (long)gridDim.x * blockDim.x;
  for (long i = (long)blockIdx.x * blockDim.x + threadIdx.x; i < TOT; i += stride) {
    if (i < NEFF) {
      int sel = (int)(i >> 11);        // 0 = Wq path, 1 = Wk path
      int r   = (int)(i & 2047);
      int d = r >> 2, h = r & 3;
      const float* W  = sel ? Wk : Wq;
      const float* wm = sel ? (w_map + DK) : w_map;
      const float* row = W + (long)d * DM + h * DK;
      float s = 0.f;
      #pragma unroll 4
      for (int j = 0; j < DK; ++j) s += row[j] * wm[j];
      (sel ? wk_eff : wq_eff)[d * 4 + h] = s;
    } else if (i < NEFF + NWV) {
      long t = i - NEFF; int n = (int)(t >> 9), k = (int)(t & 511);
      WvT[(long)n * 512 + k] = f2bf(Wv[(long)k * 512 + n]);
    } else if (i < NEFF + NWV + NFC) {
      long t = i - NEFF - NWV; int n = (int)(t >> 9), k = (int)(t & 511);
      fcT[(long)n * 512 + k] = f2bf(fc_w[(long)k * 512 + n]);
    } else if (i < NEFF + NWV + NFC + NF1) {
      long t = i - NEFF - NWV - NFC; int n = (int)(t / 640), k = (int)(t % 640);
      fc1T[(long)n * 640 + k] = f2bf(fc1_w[(long)k * 128 + n]);
    } else {
      long t = i - NEFF - NWV - NFC - NF1; int n = (int)(t >> 7), k = (int)(t & 127);
      fc2T[(long)n * 128 + k] = f2bf(fc2_w[(long)k * 128 + n]);
    }
  }
}

// ---------------------------------------------------------------------------
// Kernel 1: streaming attention. One block per batch element, 8 waves.
// Streams the 4 neighbor tensors once (HBM-roofline), computes scores vs
// wk_eff, masked softmax, writes attn, accumulates context c[4][512] (bf16).
// ---------------------------------------------------------------------------
__global__ __launch_bounds__(256)
void attn_kernel(const float* __restrict__ src, const float* __restrict__ src_t,
                 const float* __restrict__ src_p,
                 const float* __restrict__ seq, const float* __restrict__ seq_t,
                 const float* __restrict__ seq_e, const float* __restrict__ seq_p,
                 const unsigned char* __restrict__ mask,
                 const float* __restrict__ wq_eff, const float* __restrict__ wk_eff,
                 float* __restrict__ q_ws, unsigned short* __restrict__ c_ws,
                 float* __restrict__ out_attn) {
  __shared__ unsigned short Kb[NN][DM];   // 64 KB: K rows cached as bf16
  __shared__ float wkf[DM][4];            // 8 KB
  __shared__ float q_lds[DM];
  __shared__ float s_lds[NN][4];
  __shared__ float p_lds[NH][NN];
  __shared__ float qs_lds[NH];

  const int b = blockIdx.x;
  const int t = threadIdx.x;
  const int w = t >> 5, l = t & 31;

  // wk_eff -> LDS
  for (int i = t; i < DM * 4; i += 256) ((float*)wkf)[i] = wk_eff[i];

  // Build q_in = [src, 0, src_t, src_p]; stash to LDS + workspace
  {
    int d0 = t;
    float v0 = (d0 < FEATD) ? src[(long)b * FEATD + d0] : 0.f;
    int d1 = t + 256;
    float v1 = (d1 < 384) ? src_t[(long)b * FEATD + (d1 - 256)]
                          : src_p[(long)b * FEATD + (d1 - 384)];
    q_lds[d0] = v0; q_lds[d1] = v1;
    q_ws[(long)b * DM + d0] = v0;
    q_ws[(long)b * DM + d1] = v1;
  }
  __syncthreads();

  // q_score (wave 0)
  if (w == 0) {
    float acc0 = 0.f, acc1 = 0.f, acc2 = 0.f, acc3 = 0.f;
    for (int d = l; d < DM; d += 32) {
      float q = q_lds[d];
      acc0 += q * wq_eff[d * 4 + 0];
      acc1 += q * wq_eff[d * 4 + 1];
      acc2 += q * wq_eff[d * 4 + 2];
      acc3 += q * wq_eff[d * 4 + 3];
    }
    acc0 = wred(acc0); acc1 = wred(acc1); acc2 = wred(acc2); acc3 = wred(acc3);
    if (l == 0) { qs_lds[0] = acc0; qs_lds[1] = acc1; qs_lds[2] = acc2; qs_lds[3] = acc3; }
  }

  // Stream K rows; k_in = [seq, seq_e, seq_t, seq_p]. Lane l owns 16 dims.
  const float* segp[4] = { seq + (long)b * NN * FEATD, seq_e + (long)b * NN * FEATD,
                           seq_t + (long)b * NN * FEATD, seq_p + (long)b * NN * FEATD };
  const int seg   = l >> 3;          // which source tensor (16 dims fit in one)
  const int so    = (l & 7) * 16;    // offset within that tensor's 128 dims
  const int dbase = l * 16;          // position in the 512-dim concat row
  const float* segbase = segp[seg];

  for (int kr = w * 8; kr < w * 8 + 8; ++kr) {
    float v[16];
    const float* p = segbase + kr * FEATD + so;
    #pragma unroll
    for (int q4 = 0; q4 < 4; ++q4) {
      float4 f = ((const float4*)p)[q4];
      v[q4 * 4 + 0] = f.x; v[q4 * 4 + 1] = f.y; v[q4 * 4 + 2] = f.z; v[q4 * 4 + 3] = f.w;
    }
    float a0 = 0.f, a1 = 0.f, a2 = 0.f, a3 = 0.f;
    #pragma unroll
    for (int j = 0; j < 16; ++j) {
      float4 wv = *((const float4*)&wkf[dbase + j][0]);
      a0 += v[j] * wv.x; a1 += v[j] * wv.y; a2 += v[j] * wv.z; a3 += v[j] * wv.w;
    }
    unsigned short uu[16];
    #pragma unroll
    for (int j = 0; j < 16; ++j) uu[j] = f2bf(v[j]);
    *((us8*)&Kb[kr][dbase])     = *(const us8*)&uu[0];
    *((us8*)&Kb[kr][dbase + 8]) = *(const us8*)&uu[8];
    a0 = wred(a0); a1 = wred(a1); a2 = wred(a2); a3 = wred(a3);
    if (l == 0) { s_lds[kr][0] = a0; s_lds[kr][1] = a1; s_lds[kr][2] = a2; s_lds[kr][3] = a3; }
  }
  __syncthreads();

  // Masked softmax over k (waves 0..3 = heads); write attn output
  if (w < NH) {
    const int h = w;
    float qs = qs_lds[h];
    int k0 = l, k1 = l + 32;
    float s0 = qs + s_lds[k0][h];
    float s1 = qs + s_lds[k1][h];
    if (mask[(long)b * NN + k0]) s0 = -1e10f;
    if (mask[(long)b * NN + k1]) s1 = -1e10f;
    float m = wmaxr(fmaxf(s0, s1));
    float e0 = __expf(s0 - m), e1 = __expf(s1 - m);
    float sum = wred(e0 + e1);
    float inv = 1.0f / sum;
    e0 *= inv; e1 *= inv;
    p_lds[h][k0] = e0; p_lds[h][k1] = e1;
    out_attn[(long)b * (NH * NN) + h * NN + k0] = e0;
    out_attn[(long)b * (NH * NN) + h * NN + k1] = e1;
  }
  __syncthreads();

  // Context: c[h][d] = sum_k p[h,k] * K[k,d]; emit bf16 (WMMA A operand)
  {
    const int h  = t >> 6;       // 0..3
    const int dl = t & 63;       // 0..63
    float acc[8] = {0.f, 0.f, 0.f, 0.f, 0.f, 0.f, 0.f, 0.f};
    for (int k = 0; k < NN; ++k) {
      float pk = p_lds[h][k];
      #pragma unroll
      for (int j = 0; j < 8; ++j)
        acc[j] += pk * bf2f(Kb[k][dl + 64 * j]);
    }
    unsigned short* cp = c_ws + ((long)b * NH + h) * DM;
    #pragma unroll
    for (int j = 0; j < 8; ++j) cp[dl + 64 * j] = f2bf(acc[j]);
  }
}

// ---------------------------------------------------------------------------
// Kernel 2: WMMA tail. 16 batch rows per block, 8 waves.
//   out16 = per-head c @ Wv          (bf16 WMMA, f32 acc)
//   y     = LN(leaky(out16@fc_w+b) + q_in)
//   z     = relu([y,src]@fc1+b1) @ fc2 + b2
// ---------------------------------------------------------------------------
__global__ __launch_bounds__(256)
void tail_kernel(const float* __restrict__ src, const float* __restrict__ fc_b,
                 const float* __restrict__ ln_g, const float* __restrict__ ln_b,
                 const float* __restrict__ fc1_b, const float* __restrict__ fc2_b,
                 const float* __restrict__ q_ws, const unsigned short* __restrict__ c_ws,
                 const unsigned short* __restrict__ WvT, const unsigned short* __restrict__ fcT,
                 const unsigned short* __restrict__ fc1T, const unsigned short* __restrict__ fc2T,
                 float* __restrict__ out_z) {
  __shared__ unsigned short out16[16][DM];     // 16 KB bf16 attn-output
  __shared__ float          y32[16][DM];       // 32 KB post leaky+residual (f32 for LN)
  __shared__ unsigned short xA[16][XDIM];      // 20 KB LN result ++ src, bf16
  __shared__ unsigned short h1[16][FEATD];     // 4 KB

  const int b16 = blockIdx.x * 16;
  const int t = threadIdx.x, w = t >> 5, L = t & 31;
  const int lm = L & 15, lh = L >> 4;

  // ---- Stage A: out16[:, h*128..] = c_h @ Wv[:, h*128..] ----
  #pragma unroll
  for (int tt = 0; tt < 4; ++tt) {
    int ntile = w + tt * 8;              // 0..31 column tiles of out16
    int h = ntile >> 3;                  // owning head
    v8f acc = {};
    const unsigned short* cb = c_ws + ((long)b16 * NH + h) * DM;  // row m at + m*NH*DM
    #pragma unroll
    for (int kt = 0; kt < 16; ++kt) {
      v16bf a  = load_a_bf16(cb, NH * DM, kt);
      v16bf bb = load_b_wt(WvT, DM, ntile, kt);
      acc = __builtin_amdgcn_wmma_f32_16x16x32_bf16(false, a, false, bb, (short)0, acc, false, false);
    }
    int n = ntile * 16 + lm;
    #pragma unroll
    for (int r = 0; r < 8; ++r) out16[r + 8 * lh][n] = f2bf(acc[r]);
  }
  __syncthreads();

  // ---- Stage B: fc + leaky_relu(0.2) + residual(q_in) -> y32 ----
  #pragma unroll
  for (int tt = 0; tt < 4; ++tt) {
    int ntile = w + tt * 8;
    v8f acc = {};
    #pragma unroll
    for (int kt = 0; kt < 16; ++kt) {
      v16bf a  = load_a_bf16(&out16[0][0], DM, kt);
      v16bf bb = load_b_wt(fcT, DM, ntile, kt);
      acc = __builtin_amdgcn_wmma_f32_16x16x32_bf16(false, a, false, bb, (short)0, acc, false, false);
    }
    int n = ntile * 16 + lm;
    float bias = fc_b[n];
    #pragma unroll
    for (int r = 0; r < 8; ++r) {
      int m = r + 8 * lh;
      float v = acc[r] + bias;
      v = (v > 0.f) ? v : 0.2f * v;
      v += q_ws[(long)(b16 + m) * DM + n];
      y32[m][n] = v;
    }
  }
  __syncthreads();

  // ---- LayerNorm over d_model, then build xA = [LN(y)*g+b , src] bf16 ----
  #pragma unroll
  for (int rr = 0; rr < 2; ++rr) {
    int m = w * 2 + rr;
    float s = 0.f, ss = 0.f;
    #pragma unroll
    for (int j = 0; j < 16; ++j) {
      float v = y32[m][L + 32 * j];
      s += v; ss += v * v;
    }
    s = wred(s); ss = wred(ss);
    float mu   = s * (1.0f / (float)DM);
    float var  = ss * (1.0f / (float)DM) - mu * mu;
    float rstd = rsqrtf(var + 1e-5f);
    #pragma unroll
    for (int j = 0; j < 16; ++j) {
      int d = L + 32 * j;
      float v = (y32[m][d] - mu) * rstd * ln_g[d] + ln_b[d];
      xA[m][d] = f2bf(v);
    }
    #pragma unroll
    for (int j = 0; j < 4; ++j) {
      int d = L + 32 * j;
      xA[m][DM + d] = f2bf(src[(long)(b16 + m) * FEATD + d]);
    }
  }
  __syncthreads();

  // ---- Stage C: h1 = relu(xA @ fc1 + b1), 128 cols = 8 tiles (1/wave) ----
  {
    int ntile = w;
    v8f acc = {};
    #pragma unroll
    for (int kt = 0; kt < XDIM / 32; ++kt) {  // 20 k-steps
      v16bf a  = load_a_bf16(&xA[0][0], XDIM, kt);
      v16bf bb = load_b_wt(fc1T, XDIM, ntile, kt);
      acc = __builtin_amdgcn_wmma_f32_16x16x32_bf16(false, a, false, bb, (short)0, acc, false, false);
    }
    int n = ntile * 16 + lm;
    float bias = fc1_b[n];
    #pragma unroll
    for (int r = 0; r < 8; ++r) {
      float v = acc[r] + bias;
      h1[r + 8 * lh][n] = f2bf(v > 0.f ? v : 0.f);
    }
  }
  __syncthreads();

  // ---- Stage D: z = h1 @ fc2 + b2 -> d_out ----
  {
    int ntile = w;
    v8f acc = {};
    #pragma unroll
    for (int kt = 0; kt < 4; ++kt) {
      v16bf a  = load_a_bf16(&h1[0][0], FEATD, kt);
      v16bf bb = load_b_wt(fc2T, FEATD, ntile, kt);
      acc = __builtin_amdgcn_wmma_f32_16x16x32_bf16(false, a, false, bb, (short)0, acc, false, false);
    }
    int n = ntile * 16 + lm;
    float bias = fc2_b[n];
    #pragma unroll
    for (int r = 0; r < 8; ++r) {
      int m = r + 8 * lh;
      out_z[(long)(b16 + m) * FEATD + n] = acc[r] + bias;
    }
  }
}

// ---------------------------------------------------------------------------
// Launch
// ---------------------------------------------------------------------------
extern "C" void kernel_launch(void* const* d_in, const int* in_sizes, int n_in,
                              void* d_out, int out_size, void* d_ws, size_t ws_size,
                              hipStream_t stream) {
  (void)in_sizes; (void)n_in; (void)out_size; (void)ws_size;

  const float* src   = (const float*)d_in[0];
  const float* src_t = (const float*)d_in[1];
  const float* src_p = (const float*)d_in[2];
  const float* seq   = (const float*)d_in[3];
  const float* seq_t = (const float*)d_in[4];
  const float* seq_e = (const float*)d_in[5];
  const float* seq_p = (const float*)d_in[6];
  const unsigned char* mask = (const unsigned char*)d_in[7];  // jax bool = 1 byte
  const float* Wq    = (const float*)d_in[8];
  const float* Wk    = (const float*)d_in[9];
  const float* Wv    = (const float*)d_in[10];
  const float* w_map = (const float*)d_in[11];
  const float* fc_w  = (const float*)d_in[12];
  const float* fc_b  = (const float*)d_in[13];
  const float* ln_g  = (const float*)d_in[14];
  const float* ln_b  = (const float*)d_in[15];
  const float* fc1_w = (const float*)d_in[16];
  const float* fc1_b = (const float*)d_in[17];
  const float* fc2_w = (const float*)d_in[18];
  const float* fc2_b = (const float*)d_in[19];

  // Workspace layout (bytes), total ~13.2 MB
  char* ws = (char*)d_ws;
  float*          wq_eff = (float*)(ws + 0);                 //  8 KB [512][4]
  float*          wk_eff = (float*)(ws + 8192);              //  8 KB [512][4]
  float*          q_ws   = (float*)(ws + 16384);             //  4 MB [2048][512] f32
  unsigned short* c_ws   = (unsigned short*)(ws + 4210688);  //  8 MB [2048][4][512] bf16
  unsigned short* WvT    = (unsigned short*)(ws + 12599296); // 512 KB bf16 [512][512] (T)
  unsigned short* fcT    = (unsigned short*)(ws + 13123584); // 512 KB bf16 [512][512] (T)
  unsigned short* fc1T   = (unsigned short*)(ws + 13647872); // 160 KB bf16 [128][640] (T)
  unsigned short* fc2T   = (unsigned short*)(ws + 13811712); //  32 KB bf16 [128][128] (T)

  float* out_z    = (float*)d_out;                 // [B,1,128]
  float* out_attn = out_z + (long)B_TOT * FEATD;   // [B,4,1,64]

  prep_kernel<<<256, 256, 0, stream>>>(Wq, Wk, Wv, w_map, fc_w, fc1_w, fc2_w,
                                       wq_eff, wk_eff, WvT, fcT, fc1T, fc2T);

  attn_kernel<<<B_TOT, 256, 0, stream>>>(src, src_t, src_p, seq, seq_t, seq_e, seq_p,
                                         mask, wq_eff, wk_eff, q_ws, c_ws, out_attn);

  tail_kernel<<<B_TOT / 16, 256, 0, stream>>>(src, fc_b, ln_g, ln_b, fc1_b, fc2_b,
                                              q_ws, c_ws, WvT, fcT, fc1T, fc2T, out_z);
}
